// SelfAttHeadLevel4_51969104281926
// MI455X (gfx1250) — matrix operand order repaired and checked by reading, not compiled, not run on hardware
//
#include <hip/hip_runtime.h>

// ---------------------------------------------------------------------------
// Fused causal self-attention (head-level), MI455X / gfx1250, wave32 + WMMA.
// Pipeline: cast -> transpose -> QKV gemm -> QK^T scores -> softmax -> P@V.
// All matrix math: v_wmma_f32_16x16x32_bf16 (f32 accumulate).
// Double-buffered LDS tiles filled with global_load_async_to_lds_b128;
// async copy of step k+1 overlaps WMMA of step k. __launch_bounds__(256,1)
// raises the VGPR cap to keep the 128 accumulator VGPRs spill-free.
// ---------------------------------------------------------------------------

typedef __bf16 bf16_t;
typedef __attribute__((ext_vector_type(16))) __bf16 v16bf;
typedef __attribute__((ext_vector_type(8)))  float  v8f;
typedef __attribute__((ext_vector_type(4)))  unsigned int v4u;

union FragU { v16bf f; v4u u[2]; };

static constexpr int Bn   = 4;
static constexpr int S    = 2048;
static constexpr int EMB  = 2048;
static constexpr int DK   = 128;
static constexpr int BS   = Bn * S;          // 8192 rows total
static constexpr int NQKV = 2 * DK + EMB;    // 2304

// Block C tile: 128 (M) x 256 (N). 8 waves: wm 0..3 x wn 0..1,
// each wave computes 2 m-tiles x 8 n-tiles of 16x16.
static constexpr int BM = 128;
static constexpr int BN = 256;
// LDS tile row stride (bf16 elems): 32 data + 8 pad = 80B to spread rows
// across the 64 LDS banks (64B stride would alias every 4th row).
static constexpr int LDP = 40;
static constexpr int ATILE = BM * LDP;       // bf16 elems per A k-slab
static constexpr int BTILE = BN * LDP;       // bf16 elems per B k-slab

// --------------------------- WMMA wrapper ----------------------------------
__device__ __forceinline__ v8f wmma_bf16(v16bf a, v16bf b, v8f c) {
    return __builtin_amdgcn_wmma_f32_16x16x32_bf16(
        /*neg_a=*/false, a, /*neg_b=*/false, b,
        /*c_mod=*/(short)0, c, /*reuse_a=*/false, /*reuse_b=*/false);
}

// ------------------- async global -> LDS tile staging ----------------------
// Copies ROWS x 32-bf16 (64B/row) tile with per-lane b128 async transfers.
template <int ROWS>
__device__ __forceinline__ void async_tile(bf16_t* lds, const bf16_t* g,
                                           int ldg, int tid) {
    constexpr int PER = (ROWS * 4) / 256;    // 16B chunks per thread
    int c = tid * PER;
#pragma unroll
    for (int it = 0; it < PER; ++it, ++c) {
        int row  = c >> 2;
        int part = c & 3;
        const bf16_t* src = g + (size_t)row * ldg + part * 8;
        bf16_t*       dst = lds + row * LDP + part * 8;
        unsigned loff = (unsigned)(size_t)dst;                 // low 32b = LDS offset
        unsigned long long ga = (unsigned long long)(size_t)src;
        asm volatile("global_load_async_to_lds_b128 %0, %1, off"
                     :: "v"(loff), "v"(ga) : "memory");
    }
}

// --------------------- LDS fragment loads (row stride LDP) -----------------
__device__ __forceinline__ v16bf lds_a_frag(const bf16_t* tile, int mtile, int lane) {
    int m  = lane & 15;
    int hi = lane >> 4;
    const bf16_t* p = tile + (mtile * 16 + m) * LDP + hi * 8;
    FragU fr;
    fr.u[0] = *(const v4u*)p;          // K = hi*8 .. +7
    fr.u[1] = *(const v4u*)(p + 16);   // K = 16 + hi*8 .. +7
    return fr.f;
}

__device__ __forceinline__ v16bf lds_b_frag(const bf16_t* tile, int ntile, int lane) {
    int n   = lane & 15;
    int khi = (lane >> 4) * 8;
    const bf16_t* p = tile + (ntile * 16 + n) * LDP + khi;
    FragU fr;
    fr.u[0] = *(const v4u*)p;          // K = khi .. +7
    fr.u[1] = *(const v4u*)(p + 16);   // K = 16 + khi .. +7
    return fr.f;
}

// ------------------------------ GEMM core ----------------------------------
// C(BM x BN) = A(BM x kdim, row-major) * Bt(BN x kdim, row-major)^T.
// Double-buffered pipeline: async-stage slab s+1 while computing slab s.
__device__ __forceinline__ void gemm_core(const bf16_t* A, int lda,
                                          const bf16_t* Bt, int ldb,
                                          int kdim,
                                          bf16_t* tA, bf16_t* tB,   // 2x slabs each
                                          v8f acc[2][8]) {
    const int tid  = threadIdx.x;
    const int lane = tid & 31;
    const int wid  = tid >> 5;
    const int wm   = wid >> 1;           // 0..3
    const int wn   = wid & 1;            // 0..1

    // Prologue: stage slab 0 into buffer 0.
    async_tile<BM>(tA, A, lda, tid);
    async_tile<BN>(tB, Bt, ldb, tid);

    const int nsteps = kdim >> 5;
    for (int s = 0; s < nsteps; ++s) {
        // Current slab's async copies (issued last iteration) must be done in
        // every wave; the barrier also guarantees everyone finished reading
        // the buffer we are about to overwrite below.
        asm volatile("s_wait_asynccnt 0x0" ::: "memory");
        __syncthreads();

        bf16_t* cA = tA + (s & 1) * ATILE;
        bf16_t* cB = tB + (s & 1) * BTILE;

        if (s + 1 < nsteps) {            // prefetch next slab into other buffer
            int k1 = (s + 1) << 5;
            async_tile<BM>(tA + ((s + 1) & 1) * ATILE, A + k1, lda, tid);
            async_tile<BN>(tB + ((s + 1) & 1) * BTILE, Bt + k1, ldb, tid);
        }

        v16bf a0 = lds_a_frag(cA, wm * 2 + 0, lane);
        v16bf a1 = lds_a_frag(cA, wm * 2 + 1, lane);
#pragma unroll
        for (int j = 0; j < 8; ++j) {
            v16bf b = lds_b_frag(cB, wn * 8 + j, lane);
            acc[0][j] = wmma_bf16(a0, b, acc[0][j]);
            acc[1][j] = wmma_bf16(a1, b, acc[1][j]);
        }
    }
}

__device__ __forceinline__ void zero_acc(v8f acc[2][8]) {
    v8f z = {0.f, 0.f, 0.f, 0.f, 0.f, 0.f, 0.f, 0.f};
#pragma unroll
    for (int i = 0; i < 2; ++i)
#pragma unroll
        for (int j = 0; j < 8; ++j) acc[i][j] = z;
}

// C/D layout: element r of v8f -> row (lane>=16 ? 8 : 0)+r, col (lane&15).
__device__ __forceinline__ void store_tile_f32(float* C, size_t ldc, int gm0, int gn0,
                                               v8f a, int lane) {
    int col   = gn0 + (lane & 15);
    int rbase = gm0 + ((lane >> 4) << 3);
#pragma unroll
    for (int r = 0; r < 8; ++r)
        C[(size_t)(rbase + r) * ldc + col] = a[r];
}

// ------------------------------- kernels -----------------------------------

// E fp32 -> bf16, vectorized x4.
__global__ void __launch_bounds__(256) k_cast_bf16(const float* __restrict__ src,
                                                   bf16_t* __restrict__ dst,
                                                   long long n) {
    long long i = ((long long)blockIdx.x * blockDim.x + threadIdx.x) * 4;
    if (i + 3 >= n) return;
    float4 f = *(const float4*)(src + i);
    union { bf16_t h[4]; uint2 u; } o;
    o.h[0] = (bf16_t)f.x; o.h[1] = (bf16_t)f.y;
    o.h[2] = (bf16_t)f.z; o.h[3] = (bf16_t)f.w;
    *(uint2*)(dst + i) = o.u;
}

// W [EMB x NQKV] fp32 -> Wt [NQKV x EMB] bf16 (LDS-tiled transpose).
__global__ void __launch_bounds__(256) k_transpose_w(const float* __restrict__ W,
                                                     bf16_t* __restrict__ Wt) {
    __shared__ bf16_t t[32][33];
    int c0 = blockIdx.x * 32;  // over NQKV
    int r0 = blockIdx.y * 32;  // over EMB
#pragma unroll
    for (int i = 0; i < 32; i += 8) {
        int r = r0 + threadIdx.y + i;
        int c = c0 + threadIdx.x;
        t[threadIdx.y + i][threadIdx.x] = (bf16_t)W[(size_t)r * NQKV + c];
    }
    __syncthreads();
#pragma unroll
    for (int i = 0; i < 32; i += 8) {
        int n = c0 + threadIdx.y + i;
        int k = r0 + threadIdx.x;
        Wt[(size_t)n * EMB + k] = t[threadIdx.x][threadIdx.y + i];
    }
}

// QKV = Ebf @ W  (via Wt). Writes Q,K row-major [BS x DK] and V transposed
// Vt[b][n][s] so the P@V gemm gets K-contiguous B fragments.
__global__ void __launch_bounds__(256, 1)
k_qkv_gemm(const bf16_t* __restrict__ Ebf, const bf16_t* __restrict__ Wt,
           bf16_t* __restrict__ Q, bf16_t* __restrict__ K,
           bf16_t* __restrict__ Vt) {
    __shared__ bf16_t tA[2 * ATILE];
    __shared__ bf16_t tB[2 * BTILE];
    const int blockRow = blockIdx.y * BM;       // over BS
    const int blockCol = blockIdx.x * BN;       // over NQKV
    const int lane = threadIdx.x & 31;
    const int wid  = threadIdx.x >> 5;
    const int wm = wid >> 1, wn = wid & 1;

    v8f acc[2][8];
    zero_acc(acc);
    gemm_core(Ebf + (size_t)blockRow * EMB, EMB,
              Wt + (size_t)blockCol * EMB, EMB, EMB, tA, tB, acc);

#pragma unroll
    for (int i = 0; i < 2; ++i)
#pragma unroll
        for (int j = 0; j < 8; ++j) {
            int gm0 = blockRow + wm * 32 + i * 16 + ((lane >> 4) << 3);
            int gc  = blockCol + wn * 128 + j * 16 + (lane & 15);
#pragma unroll
            for (int r = 0; r < 8; ++r) {
                int grow = gm0 + r;
                bf16_t h = (bf16_t)acc[i][j][r];
                if (gc < DK) {
                    Q[(size_t)grow * DK + gc] = h;
                } else if (gc < 2 * DK) {
                    K[(size_t)grow * DK + (gc - DK)] = h;
                } else {
                    int b = grow >> 11;          // S == 2048
                    int ss = grow & (S - 1);
                    Vt[((size_t)b * EMB + (gc - 2 * DK)) * S + ss] = h;
                }
            }
        }
}

// Sc[b][q][k] = Q[b,q,:] . K[b,k,:]   (unscaled; scale applied in softmax).
// Blocks strictly above the causal diagonal are skipped entirely.
__global__ void __launch_bounds__(256, 1)
k_scores_gemm(const bf16_t* __restrict__ Q, const bf16_t* __restrict__ K,
              float* __restrict__ Sc) {
    // N block is 256 keys wide, M block 128 queries: skip iff x*256 > y*128+127.
    if (blockIdx.x * 2 > blockIdx.y) return;
    __shared__ bf16_t tA[2 * ATILE];
    __shared__ bf16_t tB[2 * BTILE];
    const int b = blockIdx.z;
    const int blockRow = blockIdx.y * BM;       // q
    const int blockCol = blockIdx.x * BN;       // k
    const int lane = threadIdx.x & 31;
    const int wid  = threadIdx.x >> 5;
    const int wm = wid >> 1, wn = wid & 1;

    const bf16_t* Qb = Q + (size_t)b * S * DK;
    const bf16_t* Kb = K + (size_t)b * S * DK;

    v8f acc[2][8];
    zero_acc(acc);
    gemm_core(Qb + (size_t)blockRow * DK, DK,
              Kb + (size_t)blockCol * DK, DK, DK, tA, tB, acc);

    float* Cb = Sc + (size_t)b * S * S;
#pragma unroll
    for (int i = 0; i < 2; ++i)
#pragma unroll
        for (int j = 0; j < 8; ++j)
            store_tile_f32(Cb, S,
                           blockRow + wm * 32 + i * 16,
                           blockCol + wn * 128 + j * 16, acc[i][j], lane);
}

// Row softmax with causal mask (k <= q), scaled by 1/sqrt(DK); P in bf16.
__global__ void __launch_bounds__(256) k_softmax(const float* __restrict__ Sc,
                                                 bf16_t* __restrict__ P) {
    const float scale = 0.08838834764831845f;   // 1/sqrt(128)
    const int row = blockIdx.x;                 // b*S + q
    const int q   = row & (S - 1);
    const float* srow = Sc + (size_t)row * S;
    bf16_t*      prow = P  + (size_t)row * S;

    const int tid  = threadIdx.x;
    const int lane = tid & 31;
    const int wid  = tid >> 5;
    __shared__ float red[8];

    float vals[8];
    float mx = -INFINITY;
#pragma unroll
    for (int i = 0; i < 8; ++i) {
        int k = tid + i * 256;
        float v = -INFINITY;
        if (k <= q) v = srow[k] * scale;
        vals[i] = v;
        mx = fmaxf(mx, v);
    }
#pragma unroll
    for (int o = 16; o > 0; o >>= 1) mx = fmaxf(mx, __shfl_xor(mx, o));
    if (lane == 0) red[wid] = mx;
    __syncthreads();
    mx = red[0];
#pragma unroll
    for (int w = 1; w < 8; ++w) mx = fmaxf(mx, red[w]);
    __syncthreads();

    float sum = 0.f;
#pragma unroll
    for (int i = 0; i < 8; ++i) {
        float e = __expf(vals[i] - mx);         // exp(-inf) == 0 for masked
        vals[i] = e;
        sum += e;
    }
#pragma unroll
    for (int o = 16; o > 0; o >>= 1) sum += __shfl_xor(sum, o);
    if (lane == 0) red[wid] = sum;
    __syncthreads();
    sum = red[0];
#pragma unroll
    for (int w = 1; w < 8; ++w) sum += red[w];
    float inv = 1.f / sum;

#pragma unroll
    for (int i = 0; i < 8; ++i) {
        int k = tid + i * 256;
        prow[k] = (bf16_t)(vals[i] * inv);      // masked entries -> exact 0
    }
}

// out[b,q,:] = P[b,q,:] @ V[b]   (B-frags from Vt; K loop clipped causally).
__global__ void __launch_bounds__(256, 1)
k_pv_gemm(const bf16_t* __restrict__ P, const bf16_t* __restrict__ Vt,
          float* __restrict__ out) {
    __shared__ bf16_t tA[2 * ATILE];
    __shared__ bf16_t tB[2 * BTILE];
    const int b = blockIdx.z;
    const int blockRow = blockIdx.y * BM;       // q
    const int blockCol = blockIdx.x * BN;       // emb
    const int lane = threadIdx.x & 31;
    const int wid  = threadIdx.x >> 5;
    const int wm = wid >> 1, wn = wid & 1;
    const int klim = (blockIdx.y + 1) * BM;     // keys beyond q-tile are all zero

    const bf16_t* Pb  = P  + (size_t)b * S * S;
    const bf16_t* Vtb = Vt + (size_t)b * EMB * S;

    v8f acc[2][8];
    zero_acc(acc);
    gemm_core(Pb + (size_t)blockRow * S, S,
              Vtb + (size_t)blockCol * S, S, klim, tA, tB, acc);

    float* Cb = out + (size_t)b * S * EMB;
#pragma unroll
    for (int i = 0; i < 2; ++i)
#pragma unroll
        for (int j = 0; j < 8; ++j)
            store_tile_f32(Cb, EMB,
                           blockRow + wm * 32 + i * 16,
                           blockCol + wn * 128 + j * 16, acc[i][j], lane);
}

// ------------------------------ launcher -----------------------------------
extern "C" void kernel_launch(void* const* d_in, const int* in_sizes, int n_in,
                              void* d_out, int out_size, void* d_ws, size_t ws_size,
                              hipStream_t stream) {
    const float* E = (const float*)d_in[0];     // [B,S,EMB]
    const float* W = (const float*)d_in[1];     // [EMB, NQKV]
    (void)in_sizes; (void)n_in; (void)out_size; // mask rebuilt from indices

    char* ws = (char*)d_ws;
    size_t offEbf = 0;
    size_t offWt  = offEbf + (size_t)BS * EMB * sizeof(bf16_t);    // 33.55 MB
    size_t offQ   = offWt  + (size_t)NQKV * EMB * sizeof(bf16_t);  // +9.44 MB
    size_t offK   = offQ   + (size_t)BS * DK * sizeof(bf16_t);     // +2.10 MB
    size_t offVt  = offK   + (size_t)BS * DK * sizeof(bf16_t);     // +2.10 MB
    size_t offSc  = offVt  + (size_t)Bn * EMB * S * sizeof(bf16_t);// +33.55 MB
    size_t offP   = offSc  + (size_t)Bn * S * S * sizeof(float);   // +67.11 MB
    (void)ws_size;                                                  // total ~181.4 MB

    bf16_t* Ebf = (bf16_t*)(ws + offEbf);
    bf16_t* Wt  = (bf16_t*)(ws + offWt);
    bf16_t* Q   = (bf16_t*)(ws + offQ);
    bf16_t* K   = (bf16_t*)(ws + offK);
    bf16_t* Vt  = (bf16_t*)(ws + offVt);
    float*  Sc  = (float*) (ws + offSc);
    bf16_t* P   = (bf16_t*)(ws + offP);
    float*  out = (float*)d_out;

    long long nE = (long long)BS * EMB;
    k_cast_bf16<<<(unsigned)((nE / 4 + 255) / 256), 256, 0, stream>>>(E, Ebf, nE);
    k_transpose_w<<<dim3(NQKV / 32, EMB / 32), dim3(32, 8), 0, stream>>>(W, Wt);
    k_qkv_gemm<<<dim3(NQKV / BN, BS / BM), 256, 0, stream>>>(Ebf, Wt, Q, K, Vt);
    k_scores_gemm<<<dim3(S / BN, S / BM, Bn), 256, 0, stream>>>(Q, K, Sc);
    k_softmax<<<Bn * S, 256, 0, stream>>>(Sc, P);
    k_pv_gemm<<<dim3(EMB / BN, S / BM, Bn), 256, 0, stream>>>(P, Vt, out);
}
